// SpatialLSTM_28561532518655
// MI455X (gfx1250) — matrix operand order repaired and checked by reading, not compiled, not run on hardware
//
#include <hip/hip_runtime.h>
#include <hip/hip_bf16.h>

typedef __attribute__((ext_vector_type(16))) __bf16 v16bf;
typedef __attribute__((ext_vector_type(8)))  float  v8f;

#define BATCH 32
#define MM    28
#define NN    28
#define HH    256
#define GCOLS 30   // grid cols (N+2)
#define ZW    1280 // 5*H

__device__ __forceinline__ float sigf(float x) {
    return 1.0f / (1.0f + __expf(-x));
}
__device__ __forceinline__ float tanhf_fast(float x) {
    // tanh(x) = 2*sigmoid(2x) - 1 ; one v_exp_f32 instead of an ocml call
    return __builtin_fmaf(2.0f, sigf(2.0f * x), -1.0f);
}

// ---------------------------------------------------------------------------
// Prep 1: E_k tables.  E[(k*256+v)][col] = sum_h emb[v][h]*gate_w[256k+h][col]
//         (+ gate_b folded into k==0).  4*256 blocks x 256 threads.
// ---------------------------------------------------------------------------
__global__ void prep_E(const float* __restrict__ emb,
                       const float* __restrict__ gw,
                       const float* __restrict__ gb,
                       float* __restrict__ E) {
    const int kv = blockIdx.x;          // 0..1023
    const int k  = kv >> 8;
    const int v  = kv & 255;
    const int tid = threadIdx.x;        // 0..255
    float acc[5] = {0.f, 0.f, 0.f, 0.f, 0.f};
    for (int h = 0; h < 256; ++h) {
        const float e = emb[v * 256 + h];
        const float* wr = gw + (size_t)(k * 256 + h) * ZW;
#pragma unroll
        for (int c = 0; c < 5; ++c) acc[c] += e * wr[c * 256 + tid];
    }
    float* er = E + (size_t)kv * ZW;
#pragma unroll
    for (int c = 0; c < 5; ++c) {
        float z = acc[c];
        if (k == 0) z += gb[c * 256 + tid];
        er[c * 256 + tid] = z;
    }
}

// ---------------------------------------------------------------------------
// Prep 2: swizzle W_lu = gate_w[1024:1536, :] (512x1280) into WMMA B-fragment
// layout (bf16):  lane = nl + 16*(ko>>4), e = ko&15,
// element addr = ((nt*16+kb)*32+lane)*16+e
// ---------------------------------------------------------------------------
__global__ void prep_wswz(const float* __restrict__ gw, __bf16* __restrict__ wswz) {
    const int idx = blockIdx.x * blockDim.x + threadIdx.x;
    if (idx >= 512 * ZW) return;
    const int K = idx / ZW;
    const int n = idx - K * ZW;
    const float v = gw[(size_t)(1024 + K) * ZW + n];
    const int nt = n >> 4, nl = n & 15;
    const int kb = K >> 5, ko = K & 31;
    const int la = nl + ((ko >> 4) << 4);
    const int e  = ko & 15;
    wswz[(size_t)(((nt * 16 + kb) * 32) + la) * 16 + e] = (__bf16)v;
}

// Prep 3: swizzle head_w (256x256) into B-fragment layout (8 K-blocks).
__global__ void prep_hwswz(const float* __restrict__ hw, __bf16* __restrict__ hwswz) {
    const int idx = blockIdx.x * blockDim.x + threadIdx.x; // 0..65535
    const int K = idx >> 8;
    const int n = idx & 255;
    const float v = hw[(size_t)K * 256 + n];
    const int nt = n >> 4, nl = n & 15;
    const int kb = K >> 5, ko = K & 31;
    const int la = nl + ((ko >> 4) << 4);
    const int e  = ko & 15;
    hwswz[(size_t)(((nt * 8 + kb) * 32) + la) * 16 + e] = (__bf16)v;
}

// ---------------------------------------------------------------------------
// Prep 4: initialize hg borders to h_init and cg borders to 0.
// Border cells: row 0 (30 cells) + rows 1..28 x cols {0,29} (56) = 86 cells.
// Grid layout: [(r*30+c)][b][hi]  (f32)
// ---------------------------------------------------------------------------
__global__ void init_borders(const float* __restrict__ h_init,
                             float* __restrict__ hg, float* __restrict__ cg) {
    const int idx = blockIdx.x * blockDim.x + threadIdx.x;
    if (idx >= 86 * BATCH * HH) return;
    const int cell = idx >> 13;          // / 8192
    const int within = idx & 8191;
    const int b = within >> 8;
    const int hi = within & 255;
    int r, c;
    if (cell < 30) { r = 0; c = cell; }
    else { const int t = cell - 30; r = 1 + (t >> 1); c = (t & 1) ? (GCOLS - 1) : 0; }
    const size_t off = ((size_t)(r * GCOLS + c) * BATCH + b) * HH + hi;
    hg[off] = h_init[hi];
    cg[off] = 0.0f;
}

// ---------------------------------------------------------------------------
// Diagonal wavefront step. One block (512 threads = 16 waves) per cell.
// Wave w owns hi in [16w,16w+16) across all 5 gate chunks:
//   nt(c) = c*16 + w ; C-frag acc[mt][c] covers (b = mt*16 + r + 8*(lane>=16),
//   hi = 16w + (lane&15)).
// ---------------------------------------------------------------------------
__global__ void __launch_bounds__(512)
diag_step(int d, int i_lo,
          const int* __restrict__ x,
          const float* __restrict__ E,
          const __bf16* __restrict__ wswz,
          const float* __restrict__ ln_scale,
          const float* __restrict__ ln_bias,
          float* __restrict__ hg, float* __restrict__ cg,
          __bf16* __restrict__ hswz) {
    __shared__ __bf16 ldsA[16384];        // 2 M-tiles x 16 K-blocks, A-frag layout
    __shared__ int   toks[BATCH][4];
    __shared__ float red_s[BATCH][16];
    __shared__ float red_q[BATCH][16];
    __shared__ float mu_s[BATCH];
    __shared__ float rs_s[BATCH];

    const int i = i_lo + blockIdx.x;
    const int j = d - i;
    const int tid = threadIdx.x;
    const int lane = tid & 31;
    const int w = tid >> 5;

    // --- neighbor tokens (token 0 at padded positions, matching jnp.pad) ---
    if (tid < BATCH * 4) {
        const int b = tid >> 2, k = tid & 3;
        int t = 0;
        if (k == 0)      { if (j >= 1)                t = x[(b * MM + i) * NN + (j - 1)]; }
        else if (k == 1) { if (i >= 1 && j >= 1)      t = x[(b * MM + (i - 1)) * NN + (j - 1)]; }
        else if (k == 2) { if (i >= 1)                t = x[(b * MM + (i - 1)) * NN + j]; }
        else             { if (i >= 1 && j + 1 < NN)  t = x[(b * MM + (i - 1)) * NN + (j + 1)]; }
        toks[b][k] = t;
    }

    // --- stage A = [h_left | h_up] (32 x 512) as bf16 A-fragments in LDS ---
    const int rL = i + 1, cL = j;       // left neighbor cell in state grid
    const int rU = i,     cU = j + 1;   // up neighbor
    const float* hl = hg + (size_t)(rL * GCOLS + cL) * BATCH * HH;
    const float* hu = hg + (size_t)(rU * GCOLS + cU) * BATCH * HH;
#pragma unroll
    for (int t = 0; t < 32; ++t) {
        const int idx = tid + t * 512;            // 0..16383
        const int b = idx >> 9;
        const int K = idx & 511;
        const float v = (K < 256) ? hl[b * HH + K] : hu[b * HH + (K - 256)];
        const int mt = b >> 4, m = b & 15;
        const int kb = K >> 5, ko = K & 31;
        const int la = m + ((ko >> 4) << 4);
        const int e  = ko & 15;
        ldsA[((size_t)((mt * 16 + kb) * 32 + la)) * 16 + e] = (__bf16)v;
    }
    __syncthreads();

    const float* clp = cg + (size_t)(rL * GCOLS + cL) * BATCH * HH;
    const float* cup = cg + (size_t)(rU * GCOLS + cU) * BATCH * HH;

    // --- prefetch gather traffic behind the WMMA loop (global_prefetch) ---
    // E rows: 32 b x 4 k x 40 lines(128B) = 5120 lines
#pragma unroll
    for (int t = 0; t < 10; ++t) {
        const int idx = tid + t * 512;            // 0..5119
        const int b   = idx / 160;
        const int rem = idx - b * 160;
        const int k   = rem / 40;
        const int seg = rem - k * 40;
        const float* p = E + (size_t)(k * 256 + toks[b][k]) * ZW + seg * 32;
        __builtin_prefetch(p, 0, 3);
    }
    // c_left / c_up rows: 2 x 256 lines
    if (tid < 256) __builtin_prefetch(clp + tid * 32, 0, 3);
    else           __builtin_prefetch(cup + (tid - 256) * 32, 0, 3);

    // --- recurrent GEMM: (32x512) @ (512x1280), bf16 WMMA, f32 accumulate ---
    // flat bases; all fragment offsets are compile-time immediates
    const __bf16* wbase = wswz + (size_t)w * 8192 + (size_t)lane * 16;
    const __bf16* abase = ldsA + lane * 16;

    const v8f vzero = {0.f, 0.f, 0.f, 0.f, 0.f, 0.f, 0.f, 0.f};
    v8f acc[2][5];
#pragma unroll
    for (int mt = 0; mt < 2; ++mt)
#pragma unroll
        for (int c = 0; c < 5; ++c) acc[mt][c] = vzero;

#pragma unroll
    for (int kb = 0; kb < 16; ++kb) {
        const v16bf a0 = *(const v16bf*)(abase + kb * 512);
        const v16bf a1 = *(const v16bf*)(abase + 8192 + kb * 512);
#pragma unroll
        for (int c = 0; c < 5; ++c) {
            const v16bf bm = *(const v16bf*)(wbase + c * 131072 + kb * 512);
            acc[0][c] = __builtin_amdgcn_wmma_f32_16x16x32_bf16(
                false, a0, false, bm, (short)0, acc[0][c], false, false);
            acc[1][c] = __builtin_amdgcn_wmma_f32_16x16x32_bf16(
                false, a1, false, bm, (short)0, acc[1][c], false, false);
        }
    }

    // --- gates: z = WMMA + gathered E rows; cell/og in registers ---
    const int hi = w * 16 + (lane & 15);
    const int halfAdd = (lane >> 4) << 3;   // 0 or 8
    const float lnsc = ln_scale[hi];
    const float lnbi = ln_bias[hi];

    float cellv[2][8], ogv[2][8];
#pragma unroll
    for (int mt = 0; mt < 2; ++mt) {
#pragma unroll
        for (int r = 0; r < 8; ++r) {
            const int b = mt * 16 + r + halfAdd;
            const int t0 = toks[b][0], t1 = toks[b][1], t2 = toks[b][2], t3 = toks[b][3];
            float zf[5];
#pragma unroll
            for (int c = 0; c < 5; ++c) {
                const int col = c * 256 + hi;
                float z = acc[mt][c][r];
                z += E[(size_t)(0 * 256 + t0) * ZW + col];
                z += E[(size_t)(1 * 256 + t1) * ZW + col];
                z += E[(size_t)(2 * 256 + t2) * ZW + col];
                z += E[(size_t)(3 * 256 + t3) * ZW + col];
                zf[c] = z;
            }
            const float cl = clp[b * HH + hi];
            const float cu = cup[b * HH + hi];
            const float cell = sigf(zf[0]) * cl + sigf(zf[1]) * cu
                             + sigf(zf[2]) * tanhf_fast(zf[4]);
            cellv[mt][r] = cell;
            ogv[mt][r]   = zf[3];
        }
    }

    // --- LayerNorm reductions over hi (256) per batch row ---
#pragma unroll
    for (int mt = 0; mt < 2; ++mt) {
#pragma unroll
        for (int r = 0; r < 8; ++r) {
            float s = cellv[mt][r];
            float q = s * s;
#pragma unroll
            for (int m2 = 1; m2 < 16; m2 <<= 1) {
                s += __shfl_xor(s, m2, 32);
                q += __shfl_xor(q, m2, 32);
            }
            const int b = mt * 16 + r + halfAdd;
            if ((lane & 15) == 0) { red_s[b][w] = s; red_q[b][w] = q; }
        }
    }
    __syncthreads();
    if (tid < BATCH) {
        float s = 0.f, q = 0.f;
#pragma unroll
        for (int k2 = 0; k2 < 16; ++k2) { s += red_s[tid][k2]; q += red_q[tid][k2]; }
        const float mu  = s * (1.0f / 256.0f);
        const float var = q * (1.0f / 256.0f) - mu * mu;
        mu_s[tid] = mu;
        rs_s[tid] = rsqrtf(var + 1e-6f);
    }
    __syncthreads();

    // --- state, write-back (f32 grids + bf16 A-fragment layout for head) ---
    float* hcur = hg + (size_t)((i + 1) * GCOLS + (j + 1)) * BATCH * HH;
    float* ccur = cg + (size_t)((i + 1) * GCOLS + (j + 1)) * BATCH * HH;
#pragma unroll
    for (int mt = 0; mt < 2; ++mt) {
#pragma unroll
        for (int r = 0; r < 8; ++r) {
            const int b = mt * 16 + r + halfAdd;
            const float cell = cellv[mt][r];
            const float nrm = (cell - mu_s[b]) * rs_s[b] * lnsc + lnbi;
            const float st = sigf(ogv[mt][r]) * tanhf_fast(nrm);
            hcur[b * HH + hi] = st;
            ccur[b * HH + hi] = cell;
            const int s = b * (MM * NN) + i * NN + j;
            const int tile = s >> 4, m = s & 15;
            const int kb = hi >> 5, ko = hi & 31;
            const int la = m + ((ko >> 4) << 4);
            const int e  = ko & 15;
            hswz[((size_t)((tile * 8 + kb) * 32 + la)) * 16 + e] = (__bf16)st;
        }
    }
}

// ---------------------------------------------------------------------------
// Head GEMM: (25088 x 256) @ (256 x 256) + bias. 8 waves/block, 1 M-tile/wave,
// 16 N-tiles x 8 K-blocks of bf16 WMMA per wave. Flat bases + immediate offs.
// ---------------------------------------------------------------------------
__global__ void __launch_bounds__(256)
head_gemm(const __bf16* __restrict__ hswz, const __bf16* __restrict__ hwswz,
          const float* __restrict__ head_b, float* __restrict__ out) {
    const int lane = threadIdx.x & 31;
    const int w = threadIdx.x >> 5;
    const int mt = blockIdx.x * 8 + w;   // 0..1567
    const v8f vzero = {0.f, 0.f, 0.f, 0.f, 0.f, 0.f, 0.f, 0.f};
    v8f acc[16];
#pragma unroll
    for (int nt = 0; nt < 16; ++nt) acc[nt] = vzero;

    const __bf16* abase = hswz + (size_t)mt * 4096 + (size_t)lane * 16;
    const __bf16* bbase = hwswz + (size_t)lane * 16;

#pragma unroll
    for (int kb = 0; kb < 8; ++kb) {
        const v16bf a = *(const v16bf*)(abase + kb * 512);
#pragma unroll
        for (int nt = 0; nt < 16; ++nt) {
            const v16bf bm = *(const v16bf*)(bbase + nt * 4096 + kb * 512);
            acc[nt] = __builtin_amdgcn_wmma_f32_16x16x32_bf16(
                false, a, false, bm, (short)0, acc[nt], false, false);
        }
    }

    const int colBase = lane & 15;
    const int rAdd = (lane >> 4) << 3;
#pragma unroll
    for (int nt = 0; nt < 16; ++nt) {
        const int col = nt * 16 + colBase;
        const float hb = head_b[col];
#pragma unroll
        for (int r = 0; r < 8; ++r) {
            const int row = mt * 16 + r + rAdd;
            out[(size_t)row * 256 + col] = acc[nt][r] + hb;
        }
    }
}

// ---------------------------------------------------------------------------
extern "C" void kernel_launch(void* const* d_in, const int* in_sizes, int n_in,
                              void* d_out, int out_size, void* d_ws, size_t ws_size,
                              hipStream_t stream) {
    const int*   x        = (const int*)  d_in[0];
    const float* emb      = (const float*)d_in[1];
    const float* gate_w   = (const float*)d_in[2];
    const float* gate_b   = (const float*)d_in[3];
    const float* ln_scale = (const float*)d_in[4];
    const float* ln_bias  = (const float*)d_in[5];
    const float* h_init   = (const float*)d_in[6];
    const float* head_w   = (const float*)d_in[7];
    const float* head_b   = (const float*)d_in[8];
    float* out = (float*)d_out;

    char* ws = (char*)d_ws;
    size_t off = 0;
    float* E = (float*)(ws + off);            off += (size_t)4 * 256 * ZW * sizeof(float);      // 5.24 MB
    float* hg = (float*)(ws + off);           off += (size_t)29 * GCOLS * BATCH * HH * sizeof(float); // 28.5 MB
    float* cg = (float*)(ws + off);           off += (size_t)29 * GCOLS * BATCH * HH * sizeof(float); // 28.5 MB
    __bf16* wswz  = (__bf16*)(ws + off);      off += (size_t)512 * ZW * sizeof(__bf16);          // 1.31 MB
    __bf16* hwswz = (__bf16*)(ws + off);      off += (size_t)256 * 256 * sizeof(__bf16);         // 0.13 MB
    __bf16* hswz  = (__bf16*)(ws + off);      off += (size_t)BATCH * MM * NN * HH * sizeof(__bf16); // 12.8 MB

    // Parallel prep (independent; ordered by stream)
    prep_E<<<4 * 256, 256, 0, stream>>>(emb, gate_w, gate_b, E);
    prep_wswz<<<(512 * ZW + 255) / 256, 256, 0, stream>>>(gate_w, wswz);
    prep_hwswz<<<(256 * 256) / 256, 256, 0, stream>>>(head_w, hwswz);
    init_borders<<<(86 * BATCH * HH + 255) / 256, 256, 0, stream>>>(h_init, hg, cg);

    // Wavefront recurrence: 55 anti-diagonals, one block per cell
    for (int d = 0; d < MM + NN - 1; ++d) {
        const int i_lo = (d > MM - 1) ? d - (NN - 1) : 0;
        const int i_hi = (d < MM - 1) ? d : MM - 1;
        const int cd = i_hi - i_lo + 1;
        diag_step<<<cd, 512, 0, stream>>>(d, i_lo, x, E, wswz, ln_scale, ln_bias,
                                          hg, cg, hswz);
    }

    // Head projection
    head_gemm<<<(BATCH * MM * NN / 16) / 8, 256, 0, stream>>>(hswz, hwswz, head_b, out);
}
